// CausalSelfAttention_21869973471784
// MI455X (gfx1250) — matrix operand order repaired and checked by reading, not compiled, not run on hardware
//
#include <hip/hip_runtime.h>
#include <math.h>

// ---------------------------------------------------------------------------
// CDNA5 (gfx1250) causal self-attention, bf16 WMMA pipeline
//   B=2, S=2048, dim=1024, heads=16, head_dim=64
// ---------------------------------------------------------------------------

#define BB 2
#define SS 2048
#define DIM 1024
#define NH 16
#define HD 64
#define QKV_N (3 * DIM)
#define NEG_MASK -1.0e30f
#define ATTN_SCALE 0.125f // 1/sqrt(64)

typedef __attribute__((ext_vector_type(16))) __bf16 v16bf;
typedef __attribute__((ext_vector_type(8)))  __bf16 v8bf;
typedef __attribute__((ext_vector_type(4)))  __bf16 v4bf;
typedef __attribute__((ext_vector_type(8)))  float  v8f;

union V16U { v16bf v; v8bf h[2]; };

// Build a 16-element bf16 A/B operand from two 16-byte chunks.
static __device__ __forceinline__ v16bf ld16(const __bf16* p0, const __bf16* p1) {
    V16U u;
    u.h[0] = *(const v8bf*)p0;
    u.h[1] = *(const v8bf*)p1;
    return u.v;
}

static __device__ __forceinline__ v8f wmma_bf16(v16bf a, v16bf b, v8f c) {
    return __builtin_amdgcn_wmma_f32_16x16x32_bf16(
        false, a, false, b, (short)0, c, false, false);
}

// gfx1250 async global->LDS copy (ASYNCcnt-tracked, ISA 08_async_tensor §4).
// GV mode: 64-bit global address in VGPR pair, per-lane LDS byte offset in VGPR.
static __device__ __forceinline__ void async_ld_b128(const __bf16* gp, __bf16* lp) {
    unsigned loff = (unsigned)(uintptr_t)lp;                 // low 32 bits = LDS offset
    unsigned long long ga = (unsigned long long)(uintptr_t)gp;
    asm volatile("global_load_async_to_lds_b128 %0, %1, off"
                 :: "v"(loff), "v"(ga) : "memory");
}
static __device__ __forceinline__ void wait_async0() {
    asm volatile("s_wait_asynccnt 0x0" ::: "memory");
}
// allow 3 in-flight (the next k-step's prefetches) while current tile is complete
static __device__ __forceinline__ void wait_async3() {
    asm volatile("s_wait_asynccnt 0x3" ::: "memory");
}

// ---------------------------------------------------------------------------
// Kernel 1: f32 -> bf16 convert (x)
// ---------------------------------------------------------------------------
__global__ __launch_bounds__(256) void k_convert(const float* __restrict__ in,
                                                 __bf16* __restrict__ out, int n4) {
    int i = blockIdx.x * blockDim.x + threadIdx.x;
    if (i >= n4) return;
    float4 v = ((const float4*)in)[i];
    v4bf o;
    o[0] = (__bf16)v.x; o[1] = (__bf16)v.y; o[2] = (__bf16)v.z; o[3] = (__bf16)v.w;
    ((v4bf*)out)[i] = o;
}

// ---------------------------------------------------------------------------
// Kernel 2: f32 [R][C] -> bf16 [C][R] transpose (weights to N-major)
// ---------------------------------------------------------------------------
__global__ __launch_bounds__(256) void k_transpose(const float* __restrict__ in,
                                                   __bf16* __restrict__ out,
                                                   int R, int C) {
    __shared__ float tile[32][33];
    int c0 = blockIdx.x * 32, r0 = blockIdx.y * 32;
    int tx = threadIdx.x, ty = threadIdx.y;
    #pragma unroll
    for (int j = ty; j < 32; j += 8)
        tile[j][tx] = in[(size_t)(r0 + j) * C + (c0 + tx)];
    __syncthreads();
    #pragma unroll
    for (int j = ty; j < 32; j += 8)
        out[(size_t)(c0 + j) * R + (r0 + tx)] = (__bf16)tile[tx][j];
}

// ---------------------------------------------------------------------------
// LDS-tiled bf16 GEMM core: block tile 64(M) x 128(N), K-step 32,
// DOUBLE-BUFFERED async global->LDS staging (prefetch next tile while
// computing current). 8 waves, each a 2x2 grid of 16x16 WMMA tiles.
// A (row-major [M,K]) and B (N-major [N,K]).
// ---------------------------------------------------------------------------
#define LDP 40            // LDS row pitch in elements (80B: 16B-aligned, bank-spread)
#define BLK_M 64
#define BLK_N 128
#define A_SZ (BLK_M * LDP)
#define B_SZ (BLK_N * LDP)

struct GemmAcc { v8f a[2][2]; };

static __device__ __forceinline__ void gemm_block(const __bf16* __restrict__ Amat,
                                                  const __bf16* __restrict__ Bmat,
                                                  int mr0, int nc0,
                                                  __bf16* ldsA, __bf16* ldsB,
                                                  GemmAcc& acc) {
    int t = threadIdx.x;
    int lane = t & 31, wid = t >> 5;
    int g = lane >> 4, ln = lane & 15;
    int mw = wid >> 2, nw = wid & 3;  // wave -> (2 m-subtiles, 2 n-subtiles)

    // staging assignments (256 lanes; 16B per A issue, 2x16B per B issue)
    const __bf16* agp = Amat + (size_t)(mr0 + (t >> 2)) * DIM + (t & 3) * 8;   // 64 rows x 32 k
    const __bf16* bgp = Bmat + (size_t)(nc0 + (t >> 1)) * DIM + (t & 1) * 16;  // 128 rows x 32 k
    __bf16* alp = ldsA + (t >> 2) * LDP + (t & 3) * 8;
    __bf16* blp = ldsB + (t >> 1) * LDP + (t & 1) * 16;

    // per-wave LDS operand offsets
    int oa0 = ((mw * 2 + 0) * 16 + ln) * LDP + 16 * g;
    int oa1 = ((mw * 2 + 1) * 16 + ln) * LDP + 16 * g;
    int ob0 = ((nw * 2 + 0) * 16 + ln) * LDP + 16 * g;
    int ob1 = ((nw * 2 + 1) * 16 + ln) * LDP + 16 * g;

    // prologue: fill buffer 0
    async_ld_b128(agp, alp);
    async_ld_b128(bgp, blp);
    async_ld_b128(bgp + 8, blp + 8);

    #pragma unroll 1
    for (int kk = 0; kk < DIM; kk += 64) {
        // ---- phase 0: prefetch kk+32 into buf1, compute buf0 (kk) ----
        async_ld_b128(agp + kk + 32, alp + A_SZ);
        async_ld_b128(bgp + kk + 32, blp + B_SZ);
        async_ld_b128(bgp + kk + 40, blp + B_SZ + 8);
        wait_async3();          // buf0 copies retired; buf1 still in flight
        __syncthreads();
        {
            const __bf16* la0 = ldsA + oa0;
            const __bf16* la1 = ldsA + oa1;
            const __bf16* lb0 = ldsB + ob0;
            const __bf16* lb1 = ldsB + ob1;
            v16bf a0 = ld16(la0, la0 + 8);
            v16bf a1 = ld16(la1, la1 + 8);
            v16bf b0 = ld16(lb0, lb0 + 8);
            v16bf b1 = ld16(lb1, lb1 + 8);
            acc.a[0][0] = wmma_bf16(a0, b0, acc.a[0][0]);
            acc.a[0][1] = wmma_bf16(a0, b1, acc.a[0][1]);
            acc.a[1][0] = wmma_bf16(a1, b0, acc.a[1][0]);
            acc.a[1][1] = wmma_bf16(a1, b1, acc.a[1][1]);
        }
        __syncthreads();        // all reads of buf0 complete

        // ---- phase 1: prefetch kk+64 into buf0, compute buf1 (kk+32) ----
        if (kk + 64 < DIM) {
            async_ld_b128(agp + kk + 64, alp);
            async_ld_b128(bgp + kk + 64, blp);
            async_ld_b128(bgp + kk + 72, blp + 8);
            wait_async3();
        } else {
            wait_async0();
        }
        __syncthreads();
        {
            const __bf16* la0 = ldsA + A_SZ + oa0;
            const __bf16* la1 = ldsA + A_SZ + oa1;
            const __bf16* lb0 = ldsB + B_SZ + ob0;
            const __bf16* lb1 = ldsB + B_SZ + ob1;
            v16bf a0 = ld16(la0, la0 + 8);
            v16bf a1 = ld16(la1, la1 + 8);
            v16bf b0 = ld16(lb0, lb0 + 8);
            v16bf b1 = ld16(lb1, lb1 + 8);
            acc.a[0][0] = wmma_bf16(a0, b0, acc.a[0][0]);
            acc.a[0][1] = wmma_bf16(a0, b1, acc.a[0][1]);
            acc.a[1][0] = wmma_bf16(a1, b0, acc.a[1][0]);
            acc.a[1][1] = wmma_bf16(a1, b1, acc.a[1][1]);
        }
        __syncthreads();        // all reads of buf1 complete
    }
}

// ---------------------------------------------------------------------------
// Kernel 3: QKV GEMM  (x_bf16 [4096,1024]) @ (wqkvT [3072,1024])
// epilogue scatters into Q,K row-major [bh,S,64] and V^T [bh,64,S]
// ---------------------------------------------------------------------------
#define QKV_NB (QKV_N / BLK_N) // 24
__global__ __launch_bounds__(256) void k_qkv_gemm(const __bf16* __restrict__ xb,
                                                  const __bf16* __restrict__ wt,
                                                  __bf16* __restrict__ Qb,
                                                  __bf16* __restrict__ Kb,
                                                  __bf16* __restrict__ Vt) {
    __shared__ __bf16 ldsA[2 * A_SZ];
    __shared__ __bf16 ldsB[2 * B_SZ];
    int mblk = blockIdx.x / QKV_NB, nblk = blockIdx.x % QKV_NB;
    int mr0 = mblk * BLK_M, nc0 = nblk * BLK_N;

    GemmAcc acc;
    #pragma unroll
    for (int mi = 0; mi < 2; mi++)
        #pragma unroll
        for (int ni = 0; ni < 2; ni++) acc.a[mi][ni] = (v8f){};

    gemm_block(xb, wt, mr0, nc0, ldsA, ldsB, acc);

    int lane = threadIdx.x & 31, wid = threadIdx.x >> 5;
    int g = lane >> 4, ln = lane & 15;
    int mw = wid >> 2, nw = wid & 3;
    int sect = nc0 >> 10;  // 0=Q 1=K 2=V (uniform per block; 1024 % 128 == 0)

    #pragma unroll
    for (int mi = 0; mi < 2; mi++) {
        int mrb = mr0 + (mw * 2 + mi) * 16;
        int bidx = mrb >> 11;
        int srow0 = (mrb & (SS - 1)) + 8 * g;
        #pragma unroll
        for (int ni = 0; ni < 2; ni++) {
            int c  = nc0 + (nw * 2 + ni) * 16 + ln;
            int cl = c & (DIM - 1);
            int h = cl >> 6, d = cl & 63;
            if (sect < 2) {
                __bf16* dst = (sect == 0) ? Qb : Kb;
                #pragma unroll
                for (int i = 0; i < 8; i++) {
                    dst[((size_t)(bidx * NH + h) * SS + srow0 + i) * HD + d] =
                        (__bf16)acc.a[mi][ni][i];
                }
            } else {
                v8bf pk;
                #pragma unroll
                for (int i = 0; i < 8; i++) pk[i] = (__bf16)acc.a[mi][ni][i];
                *(v8bf*)(Vt + ((size_t)(bidx * NH + h) * HD + d) * SS + srow0) = pk;
            }
        }
    }
}

// ---------------------------------------------------------------------------
// Kernel 4: RoPE in-place on Q and K (bf16), pairs (d, d+32) within head_dim
// ---------------------------------------------------------------------------
__global__ __launch_bounds__(256) void k_rope(__bf16* __restrict__ Qb,
                                              __bf16* __restrict__ Kb) {
    int idx = blockIdx.x * blockDim.x + threadIdx.x; // < 2^21
    int d  = idx & 31;
    int s  = (idx >> 5) & (SS - 1);
    int bh = idx >> 16;
    float freq = __expf(-((float)(2 * d) / (float)HD) * 9.210340371976184f); // ln(1e4)
    float t = (float)s * freq;
    float sn, cs;
    __sincosf(t, &sn, &cs);
    size_t base = ((size_t)bh * SS + s) * HD + d;

    float q1 = (float)Qb[base], q2 = (float)Qb[base + 32];
    Qb[base]      = (__bf16)(q1 * cs - q2 * sn);
    Qb[base + 32] = (__bf16)(q2 * cs + q1 * sn);

    float k1 = (float)Kb[base], k2 = (float)Kb[base + 32];
    Kb[base]      = (__bf16)(k1 * cs - k2 * sn);
    Kb[base + 32] = (__bf16)(k2 * cs + k1 * sn);
}

// ---------------------------------------------------------------------------
// Kernel 5: flash attention, 1 wave per 16-row Q tile, 32 K-cols / iter
// ---------------------------------------------------------------------------
__global__ __launch_bounds__(128) void k_flash(const __bf16* __restrict__ Qb,
                                               const __bf16* __restrict__ Kb,
                                               const __bf16* __restrict__ Vt,
                                               __bf16* __restrict__ attnb) {
    __shared__ __bf16 lds_p[4][16 * 40]; // per-wave 16x32 P tile, row stride 40
    int wid  = threadIdx.x >> 5;
    int lane = threadIdx.x & 31;
    int g = lane >> 4, ln = lane & 15;
    int bh    = blockIdx.x;                 // b*16 + h
    int qtile = blockIdx.y * 4 + wid;       // 0..127
    int qbase = qtile * 16;

    // Q A-operands (A layout: lane row = ln, K = e+16g (+d0))
    const __bf16* qp = Qb + ((size_t)bh * SS + qbase + ln) * HD + 16 * g;
    v16bf aq0 = ld16(qp,      qp + 8);    // d = 0..31
    v16bf aq1 = ld16(qp + 32, qp + 40);   // d = 32..63

    float m[8], lsum[8];
    v8f o[4];
    #pragma unroll
    for (int i = 0; i < 8; i++) { m[i] = -INFINITY; lsum[i] = 0.0f; }
    #pragma unroll
    for (int t = 0; t < 4; t++) o[t] = (v8f){};

    __bf16* ldsw = &lds_p[wid][0];

    for (int j32 = 0; j32 <= qbase + 15; j32 += 32) {
        // --- S = Q K^T (two 16-col sub-tiles) ---
        const __bf16* kp0 = Kb + ((size_t)bh * SS + j32 + ln) * HD + 16 * g;
        const __bf16* kp1 = kp0 + 16 * HD;
        v8f s0 = {}, s1 = {};
        s0 = wmma_bf16(aq0, ld16(kp0,      kp0 + 8),  s0);
        s0 = wmma_bf16(aq1, ld16(kp0 + 32, kp0 + 40), s0);
        s1 = wmma_bf16(aq0, ld16(kp1,      kp1 + 8),  s1);
        s1 = wmma_bf16(aq1, ld16(kp1 + 32, kp1 + 40), s1);

        // --- online softmax over 32 columns ---
        float corr[8];
        #pragma unroll
        for (int i = 0; i < 8; i++) {
            int row = qbase + i + 8 * g;
            float v0 = s0[i] * ATTN_SCALE;
            float v1 = s1[i] * ATTN_SCALE;
            if (j32 + ln > row)      v0 = NEG_MASK;
            if (j32 + 16 + ln > row) v1 = NEG_MASK;
            float mx = fmaxf(v0, v1);
            mx = fmaxf(mx, __shfl_xor(mx, 1, 16));
            mx = fmaxf(mx, __shfl_xor(mx, 2, 16));
            mx = fmaxf(mx, __shfl_xor(mx, 4, 16));
            mx = fmaxf(mx, __shfl_xor(mx, 8, 16));
            float mn = fmaxf(m[i], mx);
            float p0 = __expf(v0 - mn);
            float p1 = __expf(v1 - mn);
            float rs = p0 + p1;
            rs += __shfl_xor(rs, 1, 16);
            rs += __shfl_xor(rs, 2, 16);
            rs += __shfl_xor(rs, 4, 16);
            rs += __shfl_xor(rs, 8, 16);
            corr[i] = __expf(m[i] - mn);
            lsum[i] = lsum[i] * corr[i] + rs;
            m[i] = mn;
            // stage P (C-layout -> LDS, bf16)
            ldsw[(i + 8 * g) * 40 + ln]      = (__bf16)p0;
            ldsw[(i + 8 * g) * 40 + 16 + ln] = (__bf16)p1;
        }
        #pragma unroll
        for (int t = 0; t < 4; t++)
            #pragma unroll
            for (int i = 0; i < 8; i++) o[t][i] *= corr[i];

        // --- reload P in A layout from LDS ---
        V16U up;
        up.h[0] = *(const v8bf*)(&ldsw[ln * 40 + 16 * g]);
        up.h[1] = *(const v8bf*)(&ldsw[ln * 40 + 16 * g + 8]);
        v16bf ap = up.v;

        // --- O += P V  (V^T B-operands: contiguous along k) ---
        #pragma unroll
        for (int t = 0; t < 4; t++) {
            const __bf16* vp = Vt + ((size_t)bh * HD + 16 * t + ln) * SS + j32 + 16 * g;
            o[t] = wmma_bf16(ap, ld16(vp, vp + 8), o[t]);
        }
    }

    // epilogue: normalize, write bf16 attn output [B, S, H*64]
    int h = bh & (NH - 1), b = bh >> 4;
    #pragma unroll
    for (int t = 0; t < 4; t++) {
        #pragma unroll
        for (int i = 0; i < 8; i++) {
            int row = qbase + i + 8 * g;
            int col = h * HD + 16 * t + ln;
            attnb[((size_t)b * SS + row) * DIM + col] = (__bf16)(o[t][i] / lsum[i]);
        }
    }
}

// ---------------------------------------------------------------------------
// Kernel 6: output projection  attn_bf16 [4096,1024] @ wprojT + bias -> f32
// ---------------------------------------------------------------------------
#define PROJ_NB (DIM / BLK_N) // 8
__global__ __launch_bounds__(256) void k_proj_gemm(const __bf16* __restrict__ ab,
                                                   const __bf16* __restrict__ wt,
                                                   const float* __restrict__ bias,
                                                   float* __restrict__ out) {
    __shared__ __bf16 ldsA[2 * A_SZ];
    __shared__ __bf16 ldsB[2 * B_SZ];
    int mblk = blockIdx.x / PROJ_NB, nblk = blockIdx.x % PROJ_NB;
    int mr0 = mblk * BLK_M, nc0 = nblk * BLK_N;

    GemmAcc acc;
    #pragma unroll
    for (int mi = 0; mi < 2; mi++)
        #pragma unroll
        for (int ni = 0; ni < 2; ni++) acc.a[mi][ni] = (v8f){};

    gemm_block(ab, wt, mr0, nc0, ldsA, ldsB, acc);

    int lane = threadIdx.x & 31, wid = threadIdx.x >> 5;
    int g = lane >> 4, ln = lane & 15;
    int mw = wid >> 2, nw = wid & 3;

    #pragma unroll
    for (int mi = 0; mi < 2; mi++) {
        int mrb = mr0 + (mw * 2 + mi) * 16 + 8 * g;
        #pragma unroll
        for (int ni = 0; ni < 2; ni++) {
            int c = nc0 + (nw * 2 + ni) * 16 + ln;
            float bv = bias[c];
            #pragma unroll
            for (int i = 0; i < 8; i++)
                out[(size_t)(mrb + i) * DIM + c] = acc.a[mi][ni][i] + bv;
        }
    }
}

// ---------------------------------------------------------------------------
// Host launcher
// ---------------------------------------------------------------------------
extern "C" void kernel_launch(void* const* d_in, const int* in_sizes, int n_in,
                              void* d_out, int out_size, void* d_ws, size_t ws_size,
                              hipStream_t stream) {
    const float* x      = (const float*)d_in[0];
    const float* w_qkv  = (const float*)d_in[1];
    const float* w_proj = (const float*)d_in[2];
    const float* b_proj = (const float*)d_in[3];
    float* out = (float*)d_out;

    const size_t MB = 1024 * 1024;
    char* ws = (char*)d_ws;
    __bf16* xb     = (__bf16*)(ws + 0);        //  8 MB: x bf16 [4096,1024]
    __bf16* wqkvt  = (__bf16*)(ws + 8  * MB);  //  6 MB: w_qkv^T bf16 [3072,1024]
    __bf16* wprojt = (__bf16*)(ws + 14 * MB);  //  2 MB: w_proj^T bf16 [1024,1024]
    __bf16* Qb     = (__bf16*)(ws + 16 * MB);  //  8 MB: Q [bh,S,64]
    __bf16* Kb     = (__bf16*)(ws + 24 * MB);  //  8 MB: K [bh,S,64]
    __bf16* Vt     = (__bf16*)(ws + 32 * MB);  //  8 MB: V^T [bh,64,S]
    __bf16* attnb  = (__bf16*)(ws + 40 * MB);  //  8 MB: attn out bf16 [B,S,1024]

    // 1) convert x -> bf16
    int n4 = BB * SS * DIM / 4; // 1M
    k_convert<<<n4 / 256, 256, 0, stream>>>(x, xb, n4);

    // 2) transpose weights to N-major bf16
    k_transpose<<<dim3(QKV_N / 32, DIM / 32), dim3(32, 8), 0, stream>>>(w_qkv, wqkvt, DIM, QKV_N);
    k_transpose<<<dim3(DIM / 32,  DIM / 32), dim3(32, 8), 0, stream>>>(w_proj, wprojt, DIM, DIM);

    // 3) QKV GEMM: 64x24 blocks of 64x128, double-buffered async-LDS staging
    k_qkv_gemm<<<(BB * SS / BLK_M) * QKV_NB, 256, 0, stream>>>(xb, wqkvt, Qb, Kb, Vt);

    // 4) RoPE on Q, K (2^21 pair threads)
    k_rope<<<(1 << 21) / 256, 256, 0, stream>>>(Qb, Kb);

    // 5) flash attention: grid (bh=32, qtile-groups=32), 4 waves/block
    k_flash<<<dim3(BB * NH, (SS / 16) / 4), 128, 0, stream>>>(Qb, Kb, Vt, attnb);

    // 6) projection GEMM + bias: 64x8 blocks
    k_proj_gemm<<<(BB * SS / BLK_M) * PROJ_NB, 256, 0, stream>>>(attnb, wprojt, b_proj, out);
}
